// MultiHeadAttentionWithCaching_42640435314893
// MI455X (gfx1250) — compile-verified, matrix operand order checked by reading, not run on hardware
//
#include <hip/hip_runtime.h>

#define D_MODEL 1024
#define SEQ     2048
#define BATCH   2
#define HEADS   16
#define DHEAD   64
#define MTOT    (BATCH * SEQ)   // 4096 rows

typedef __attribute__((ext_vector_type(16))) _Float16 v16h;
typedef __attribute__((ext_vector_type(8)))  _Float16 v8h;
typedef __attribute__((ext_vector_type(8)))  float    v8f;
typedef __attribute__((ext_vector_type(4)))  int      v4i;

// ---------------------------------------------------------------------------
// CDNA5 async global->LDS copy (ASYNCcnt-tracked), with portable fallback
// ---------------------------------------------------------------------------
#define AS1 __attribute__((address_space(1)))
#define AS3 __attribute__((address_space(3)))

#if __has_builtin(__builtin_amdgcn_global_load_async_to_lds_b128)
#define HAVE_ASYNC 1
static __device__ __forceinline__ void cp16_g2l(void* ldst, const void* gsrc) {
  // emits global_load_async_to_lds_b128 (ASYNCcnt)
  __builtin_amdgcn_global_load_async_to_lds_b128((AS1 v4i*)(void*)gsrc,
                                                 (AS3 v4i*)ldst, 0, 0);
}
#else
#define HAVE_ASYNC 0
static __device__ __forceinline__ void cp16_g2l(void* ldst, const void* gsrc) {
  *(v8h*)ldst = *(const v8h*)gsrc;   // global_load_b128 + ds_store_b128
}
#endif

#if HAVE_ASYNC
#if __has_builtin(__builtin_amdgcn_s_wait_asynccnt)
#define WAIT_ASYNC(n) __builtin_amdgcn_s_wait_asynccnt(n)
#else
#define WAIT_ASYNC(n) asm volatile("s_wait_asynccnt %0" ::"i"(n) : "memory")
#endif
#else
#define WAIT_ASYNC(n)
#endif

// ---------------------------------------------------------------------------
// WMMA helpers
// ---------------------------------------------------------------------------
static __device__ __forceinline__ v8f wmma_f16(v16h a, v16h b, v8f c) {
  // D = A(16x32 f16) * B(32x16 f16) + C(16x16 f32)
  return __builtin_amdgcn_wmma_f32_16x16x32_f16(false, a, false, b, (short)0, c,
                                                false, false);
}

// Load an A/B fragment for v_wmma_f32_16x16x32_f16 from row-major f16 memory
// (global or LDS; address space is inferred per call site).
// ISA 7.12.2: lane = {g=lane/16, m=lane%16}; the 16 halfs are two contiguous
// 8-half chunks at k = g*8 and k = 16+g*8.
static __device__ __forceinline__ v16h load_frag(const _Float16* __restrict__ base,
                                                 int row0, int ld, int kbase,
                                                 int lane) {
  int m = lane & 15, g = lane >> 4;
  const _Float16* p = base + (size_t)(row0 + m) * (size_t)ld + (size_t)(kbase + g * 8);
  v8h lo = *(const v8h*)p;
  v8h hi = *(const v8h*)(p + 16);
  v16h r;
#pragma unroll
  for (int i = 0; i < 8; ++i) { r[i] = lo[i]; r[i + 8] = hi[i]; }
  return r;
}

// ---------------------------------------------------------------------------
// Preprocessing kernels
// ---------------------------------------------------------------------------
__global__ void cvt_f32_to_f16(const float* __restrict__ in,
                               _Float16* __restrict__ out, int n) {
  int i = blockIdx.x * blockDim.x + threadIdx.x;
  if (i < n) out[i] = (_Float16)in[i];
}

// out[n*D + k] = (f16) in[k*D + n]   (D = 1024)
__global__ __launch_bounds__(1024) void transpose_w(const float* __restrict__ in,
                                                    _Float16* __restrict__ out) {
  __shared__ float tile[32][33];
  int tx = threadIdx.x, ty = threadIdx.y;
  int x0 = blockIdx.x * 32, y0 = blockIdx.y * 32;
  tile[ty][tx] = in[(size_t)(y0 + ty) * D_MODEL + (x0 + tx)];
  __syncthreads();
  out[(size_t)(x0 + ty) * D_MODEL + (y0 + tx)] = (_Float16)tile[tx][ty];
}

// ---------------------------------------------------------------------------
// WMMA GEMM:  C[4096 x 1024] = A[4096 x 1024] * B, with B given transposed
// (Bt[n][k], row-major).  Block = 256 thr = 8 waves, tile 128x128.
// Each wave: 32(M) x 64(N) = 2x4 wmma accumulators, k-step 32.
// mode 0: store Q  f16 [B,H,S,Dh], pre-scaled by 1/sqrt(Dh)
// mode 1: store K  f16 [B,H,S,Dh]
// mode 2: store V^T f16 [B,H,Dh,S]
// mode 3: store f32 row-major + bias (output projection)
// ---------------------------------------------------------------------------
__global__ __launch_bounds__(256) void gemm_wmma(const _Float16* __restrict__ A,
                                                 const _Float16* __restrict__ Bt,
                                                 void* __restrict__ out,
                                                 const float* __restrict__ bias,
                                                 int mode) {
  int tid  = threadIdx.x;
  int lane = tid & 31;
  int wave = tid >> 5;
  int wm = wave & 3;        // 4 waves along M
  int wn = wave >> 2;       // 2 waves along N
  int row0 = blockIdx.x * 128 + wm * 32;
  int col0 = blockIdx.y * 128 + wn * 64;

  v8f acc[2][4];
#pragma unroll
  for (int i = 0; i < 2; ++i)
#pragma unroll
    for (int j = 0; j < 4; ++j)
#pragma unroll
      for (int r = 0; r < 8; ++r) acc[i][j][r] = 0.0f;

  for (int k = 0; k < D_MODEL; k += 32) {
    if (k + 32 < D_MODEL) {
      __builtin_prefetch((const void*)(A  + (size_t)(row0 + (lane & 15)) * D_MODEL + k + 32), 0, 3);
      __builtin_prefetch((const void*)(Bt + (size_t)(col0 + (lane & 15)) * D_MODEL + k + 32), 0, 3);
    }
    v16h a0 = load_frag(A, row0,      D_MODEL, k, lane);
    v16h a1 = load_frag(A, row0 + 16, D_MODEL, k, lane);
    v16h b0 = load_frag(Bt, col0,      D_MODEL, k, lane);
    v16h b1 = load_frag(Bt, col0 + 16, D_MODEL, k, lane);
    v16h b2 = load_frag(Bt, col0 + 32, D_MODEL, k, lane);
    v16h b3 = load_frag(Bt, col0 + 48, D_MODEL, k, lane);
    acc[0][0] = wmma_f16(a0, b0, acc[0][0]);
    acc[0][1] = wmma_f16(a0, b1, acc[0][1]);
    acc[0][2] = wmma_f16(a0, b2, acc[0][2]);
    acc[0][3] = wmma_f16(a0, b3, acc[0][3]);
    acc[1][0] = wmma_f16(a1, b0, acc[1][0]);
    acc[1][1] = wmma_f16(a1, b1, acc[1][1]);
    acc[1][2] = wmma_f16(a1, b2, acc[1][2]);
    acc[1][3] = wmma_f16(a1, b3, acc[1][3]);
  }

  int m = lane & 15, g = lane >> 4;
#pragma unroll
  for (int i = 0; i < 2; ++i) {
#pragma unroll
    for (int j = 0; j < 4; ++j) {
#pragma unroll
      for (int r = 0; r < 8; ++r) {
        int row = row0 + i * 16 + 8 * g + r;   // C layout: M = 8*(lane/16)+vgpr
        int col = col0 + j * 16 + m;           //           N = lane%16
        float v = acc[i][j][r];
        if (mode == 3) {
          ((float*)out)[(size_t)row * D_MODEL + col] = v + bias[col];
        } else {
          int b = row >> 11, s = row & (SEQ - 1);
          int h = col >> 6,  d = col & (DHEAD - 1);
          _Float16* o = (_Float16*)out;
          if (mode == 0)
            o[(((size_t)(b * HEADS + h)) * SEQ + s) * DHEAD + d] = (_Float16)(v * 0.125f);
          else if (mode == 1)
            o[(((size_t)(b * HEADS + h)) * SEQ + s) * DHEAD + d] = (_Float16)v;
          else
            o[(((size_t)(b * HEADS + h)) * DHEAD + d) * SEQ + s] = (_Float16)v;
        }
      }
    }
  }
}

// ---------------------------------------------------------------------------
// Flash attention (causal, online softmax), LDS-shared K/V.
// Block = 256 thr = 8 waves = 128 query rows of one (b,h); each wave owns 16
// rows.  K/V chunks of 32 kv positions are staged cooperatively in LDS
// (async global->LDS, double-buffered); all 8 waves consume each chunk.
// Per chunk & wave:  S(16x32) = Q(16x64)*K^T  -> 4 wmma
//                    O += softmax(S) * V      -> 4 wmma (P transposed via LDS)
// Q pre-scaled by 1/sqrt(Dh).  V stored transposed [Dh, S].
// ---------------------------------------------------------------------------
__global__ __launch_bounds__(256) void attn_wmma(const _Float16* __restrict__ Q,
                                                 const _Float16* __restrict__ Km,
                                                 const _Float16* __restrict__ Vt,
                                                 _Float16* __restrict__ ctx) {
  __shared__ alignas(16) _Float16 kbuf[2][32 * 64];   // [kv 32][d 64]
  __shared__ alignas(16) _Float16 vbuf[2][64 * 32];   // [d 64][kv 32]
  __shared__ alignas(16) _Float16 pbuf[8][16 * 32];   // per-wave P tile

  int tid  = threadIdx.x;
  int lane = tid & 31;
  int wave = tid >> 5;
  int m = lane & 15, g = lane >> 4;

  int bh = blockIdx.y;            // 0..31  (b*16 + h)
  int b  = bh >> 4, h = bh & 15;
  int qb0   = blockIdx.x * 128;
  int qbase = qb0 + wave * 16;

  const _Float16* qh = Q  + (size_t)bh * SEQ * DHEAD;
  const _Float16* kh = Km + (size_t)bh * SEQ * DHEAD;
  const _Float16* vh = Vt + (size_t)bh * DHEAD * SEQ;

  v16h aq0 = load_frag(qh, qbase, DHEAD, 0,  lane);
  v16h aq1 = load_frag(qh, qbase, DHEAD, 32, lane);

  v8f o[4];
  float rmax[8], rsum[8];
#pragma unroll
  for (int r = 0; r < 8; ++r) {
    rmax[r] = -3.0e38f; rsum[r] = 0.0f;
    o[0][r] = 0.0f; o[1][r] = 0.0f; o[2][r] = 0.0f; o[3][r] = 0.0f;
  }

  _Float16* pl = pbuf[wave];
  const int nchunks = (qb0 + 128) / 32;   // causal: kv < qb0+128

  // cooperative-fill addressing: 256 threads x 16B cover each 4KB tile
  int krow = tid >> 3, kcol = (tid & 7) * 8;   // K: 32 rows x 64 halfs
  int vrow = tid >> 2, vcol = (tid & 3) * 8;   // V: 64 rows x 32 halfs

  // prologue: chunk 0 -> buffer 0
  cp16_g2l(&kbuf[0][krow * 64 + kcol], kh + (size_t)krow * DHEAD + kcol);
  cp16_g2l(&vbuf[0][vrow * 32 + vcol], vh + (size_t)vrow * SEQ + vcol);

  for (int c = 0; c < nchunks; ++c) {
    int kvb = c * 32;
    int cur = c & 1;
    if (c + 1 < nchunks) {
      int nb = cur ^ 1, kvn = kvb + 32;
      cp16_g2l(&kbuf[nb][krow * 64 + kcol],
               kh + (size_t)(kvn + krow) * DHEAD + kcol);
      cp16_g2l(&vbuf[nb][vrow * 32 + vcol],
               vh + (size_t)vrow * SEQ + kvn + vcol);
      WAIT_ASYNC(2);     // retire chunk c (2 newest still in flight)
    } else {
      WAIT_ASYNC(0);
    }
    __syncthreads();

    if (kvb < qbase + 16) {        // wave-uniform: skip fully-masked chunks
      const _Float16* kb = kbuf[cur];
      const _Float16* vb = vbuf[cur];

      // ---- scores: two 16x16 tiles over kv columns [kvb, kvb+32)
      v16h bk00 = load_frag(kb, 0,  64, 0,  lane);
      v16h bk01 = load_frag(kb, 0,  64, 32, lane);
      v16h bk10 = load_frag(kb, 16, 64, 0,  lane);
      v16h bk11 = load_frag(kb, 16, 64, 32, lane);
      v8f s0, s1;
#pragma unroll
      for (int r = 0; r < 8; ++r) { s0[r] = 0.0f; s1[r] = 0.0f; }
      s0 = wmma_f16(aq0, bk00, s0);
      s0 = wmma_f16(aq1, bk01, s0);
      s1 = wmma_f16(aq0, bk10, s1);
      s1 = wmma_f16(aq1, bk11, s1);

      // ---- causal mask + online softmax
#pragma unroll
      for (int r = 0; r < 8; ++r) {
        int row = qbase + 8 * g + r;
        int c0  = kvb + m;
        int c1  = kvb + 16 + m;
        float x0 = (c0 <= row) ? s0[r] : -3.0e38f;
        float x1 = (c1 <= row) ? s1[r] : -3.0e38f;
        float mx = fmaxf(x0, x1);
#pragma unroll
        for (int off = 1; off < 16; off <<= 1)
          mx = fmaxf(mx, __shfl_xor(mx, off, 32));
        float mnew = fmaxf(rmax[r], mx);
        float corr = __expf(rmax[r] - mnew);
        float p0 = __expf(x0 - mnew);
        float p1 = __expf(x1 - mnew);
        float ps = p0 + p1;
#pragma unroll
        for (int off = 1; off < 16; off <<= 1)
          ps += __shfl_xor(ps, off, 32);
        rsum[r] = rsum[r] * corr + ps;
        rmax[r] = mnew;
        o[0][r] *= corr; o[1][r] *= corr; o[2][r] *= corr; o[3][r] *= corr;
        pl[(8 * g + r) * 32 + m]      = (_Float16)p0;
        pl[(8 * g + r) * 32 + 16 + m] = (_Float16)p1;
      }

      // wave-local LDS ordering for the P transpose staging
      asm volatile("s_wait_dscnt 0" ::: "memory");
      v16h pa;
      {
        const v8h* pp = (const v8h*)&pl[m * 32 + g * 8];
        v8h lo = pp[0];
        v8h hi = pp[2];   // +16 halfs
#pragma unroll
        for (int i = 0; i < 8; ++i) { pa[i] = lo[i]; pa[i + 8] = hi[i]; }
      }
      asm volatile("s_wait_dscnt 0" ::: "memory");

      // ---- O += P * V  (V fragments from LDS, k = chunk-local kv)
      v16h bv0 = load_frag(vb, 0,  32, 0, lane);
      v16h bv1 = load_frag(vb, 16, 32, 0, lane);
      v16h bv2 = load_frag(vb, 32, 32, 0, lane);
      v16h bv3 = load_frag(vb, 48, 32, 0, lane);
      o[0] = wmma_f16(pa, bv0, o[0]);
      o[1] = wmma_f16(pa, bv1, o[1]);
      o[2] = wmma_f16(pa, bv2, o[2]);
      o[3] = wmma_f16(pa, bv3, o[3]);
    }
    __syncthreads();   // protect buffers before next issue overwrites
  }

  // ---- normalize and store ctx [B, S, D] row-major (f16)
#pragma unroll
  for (int j = 0; j < 4; ++j) {
#pragma unroll
    for (int r = 0; r < 8; ++r) {
      int row = qbase + 8 * g + r;
      int d   = h * DHEAD + j * 16 + m;
      float v = o[j][r] / rsum[r];
      ctx[((size_t)b * SEQ + row) * D_MODEL + d] = (_Float16)v;
    }
  }
}

// ---------------------------------------------------------------------------
// Host-side launcher
// ---------------------------------------------------------------------------
extern "C" void kernel_launch(void* const* d_in, const int* in_sizes, int n_in,
                              void* d_out, int out_size, void* d_ws, size_t ws_size,
                              hipStream_t stream) {
  (void)in_sizes; (void)n_in; (void)out_size; (void)ws_size;
  const float* x  = (const float*)d_in[0];
  const float* Wq = (const float*)d_in[1];
  const float* Wk = (const float*)d_in[2];
  const float* Wv = (const float*)d_in[3];
  const float* Wo = (const float*)d_in[4];
  const float* bo = (const float*)d_in[5];

  char* ws = (char*)d_ws;
  const size_t MB = 1024 * 1024;
  _Float16* Xh  = (_Float16*)(ws + 0);        // 8 MB  x in f16
  _Float16* Wqt = (_Float16*)(ws + 8 * MB);   // 2 MB  Wq^T f16
  _Float16* Wkt = (_Float16*)(ws + 10 * MB);  // 2 MB
  _Float16* Wvt = (_Float16*)(ws + 12 * MB);  // 2 MB
  _Float16* Wot = (_Float16*)(ws + 14 * MB);  // 2 MB
  _Float16* Qh  = (_Float16*)(ws + 16 * MB);  // 8 MB  [B,H,S,Dh] (pre-scaled)
  _Float16* Kh  = (_Float16*)(ws + 24 * MB);  // 8 MB  [B,H,S,Dh]
  _Float16* Vth = (_Float16*)(ws + 32 * MB);  // 8 MB  [B,H,Dh,S]
  _Float16* Ctx = (_Float16*)(ws + 40 * MB);  // 8 MB  [B,S,D]

  int nx = MTOT * D_MODEL;
  cvt_f32_to_f16<<<dim3((nx + 255) / 256), dim3(256), 0, stream>>>(x, Xh, nx);

  dim3 tg(D_MODEL / 32, D_MODEL / 32), tb(32, 32);
  transpose_w<<<tg, tb, 0, stream>>>(Wq, Wqt);
  transpose_w<<<tg, tb, 0, stream>>>(Wk, Wkt);
  transpose_w<<<tg, tb, 0, stream>>>(Wv, Wvt);
  transpose_w<<<tg, tb, 0, stream>>>(Wo, Wot);

  dim3 gg(MTOT / 128, D_MODEL / 128), gb(256);
  gemm_wmma<<<gg, gb, 0, stream>>>(Xh, Wqt, (void*)Qh,  nullptr, 0);
  gemm_wmma<<<gg, gb, 0, stream>>>(Xh, Wkt, (void*)Kh,  nullptr, 1);
  gemm_wmma<<<gg, gb, 0, stream>>>(Xh, Wvt, (void*)Vth, nullptr, 2);

  attn_wmma<<<dim3(SEQ / 128, BATCH * HEADS), dim3(256), 0, stream>>>(Qh, Kh, Vth, Ctx);

  gemm_wmma<<<gg, gb, 0, stream>>>(Ctx, Wot, d_out, bo, 3);
}
